// SFI_Selector_42099269435826
// MI455X (gfx1250) — matrix operand order repaired and checked by reading, not compiled, not run on hardware
//
#include <hip/hip_runtime.h>
#include <stdint.h>

// ---------------- problem constants (from reference) ----------------
#define KSEL 5
#define B_   64
#define C_   5
#define H_   50
#define S_   32
#define E_   768
#define D_   768
#define MROWS (B_*C_ + B_*H_)       // 3520 projected rows (cdd then his)
#define MT_   (MROWS/32)            // 110 m-tiles of 32
#define NT_   (D_/32)               // 24  n-tiles of 32

// ---------------- WMMA vector types (gfx1250, wave32) ----------------
typedef __bf16        v16bf __attribute__((ext_vector_type(16)));
typedef float         v8f   __attribute__((ext_vector_type(8)));
typedef unsigned int  u32x4 __attribute__((ext_vector_type(4)));
typedef float         f32x4 __attribute__((ext_vector_type(4)));

union Frag16 { v16bf v; u32x4 q[2]; };

// ---------------- 1) f32 -> (bf16 hi, bf16 lo) split (RNE) ----------------
__global__ __launch_bounds__(256)
void split_bf16_kernel(const float* __restrict__ src,
                       uint16_t* __restrict__ hi, uint16_t* __restrict__ lo, int n) {
  int i = blockIdx.x * 256 + threadIdx.x;
  if (i >= n) return;
  float f = src[i];
  uint32_t x  = __float_as_uint(f);
  uint32_t rh = x + 0x7FFFu + ((x >> 16) & 1u);       // round-to-nearest-even
  uint16_t h  = (uint16_t)(rh >> 16);
  float hf    = __uint_as_float(((uint32_t)h) << 16);
  float l     = f - hf;
  uint32_t xl = __float_as_uint(l);
  uint32_t rl = xl + 0x7FFFu + ((xl >> 16) & 1u);
  hi[i] = h;
  lo[i] = (uint16_t)(rl >> 16);
}

// ---------------- 2) projection GEMM: P = X @ W^T + b (split-bf16 WMMA) ----
// One wave -> 32x32 tile of P. 4 waves / block. Exact tiling (3520x768, K=768).
#define WMMA_BF16(A, Bf, Cc) \
  (Cc) = __builtin_amdgcn_wmma_f32_16x16x32_bf16(false, (A).v, false, (Bf).v, (short)0, (Cc), false, false)

__global__ __launch_bounds__(128)
void proj_wmma_kernel(const uint16_t* __restrict__ Xh, const uint16_t* __restrict__ Xl,
                      const uint16_t* __restrict__ Wh, const uint16_t* __restrict__ Wl,
                      const float* __restrict__ bias, float* __restrict__ P) {
  const int lane = threadIdx.x & 31;
  const int wid  = blockIdx.x * 4 + (threadIdx.x >> 5);  // 0..2639
  const int mt   = wid % MT_;
  const int nt   = wid / MT_;
  const int m0   = mt * 32;
  const int n0   = nt * 32;

  // A fragment addressing: lane holds row (lane&15); K-half by lane[4]
  const int rA    = lane & 15;
  const int koffA = (lane >> 4) * 8;
  // B fragment addressing: lane holds col N=(lane&15); K-half by lane[4]
  const int rB    = lane & 15;
  const int koffB = (lane >> 4) * 8;

  v8f acc00 = {}, acc01 = {}, acc10 = {}, acc11 = {};

  for (int kb = 0; kb < D_; kb += 32) {
    Frag16 ah0, ah1, al0, al1, bh0, bh1, bl0, bl1;

    const size_t a0 = (size_t)(m0 + rA)      * D_ + kb + koffA;
    const size_t a1 = (size_t)(m0 + 16 + rA) * D_ + kb + koffA;
    ah0.q[0] = *(const u32x4*)(Xh + a0);  ah0.q[1] = *(const u32x4*)(Xh + a0 + 16);
    al0.q[0] = *(const u32x4*)(Xl + a0);  al0.q[1] = *(const u32x4*)(Xl + a0 + 16);
    ah1.q[0] = *(const u32x4*)(Xh + a1);  ah1.q[1] = *(const u32x4*)(Xh + a1 + 16);
    al1.q[0] = *(const u32x4*)(Xl + a1);  al1.q[1] = *(const u32x4*)(Xl + a1 + 16);

    // B[k,n] = W[n,k]: W rows are already K-contiguous for the B fragment
    const size_t b0 = (size_t)(n0 + rB)      * D_ + kb + koffB;
    const size_t b1 = (size_t)(n0 + 16 + rB) * D_ + kb + koffB;
    bh0.q[0] = *(const u32x4*)(Wh + b0);  bh0.q[1] = *(const u32x4*)(Wh + b0 + 16);
    bl0.q[0] = *(const u32x4*)(Wl + b0);  bl0.q[1] = *(const u32x4*)(Wl + b0 + 16);
    bh1.q[0] = *(const u32x4*)(Wh + b1);  bh1.q[1] = *(const u32x4*)(Wh + b1 + 16);
    bl1.q[0] = *(const u32x4*)(Wl + b1);  bl1.q[1] = *(const u32x4*)(Wl + b1 + 16);

    // split-bf16 f32 emulation: hi*hi + hi*lo + lo*hi
    WMMA_BF16(ah0, bh0, acc00); WMMA_BF16(ah0, bl0, acc00); WMMA_BF16(al0, bh0, acc00);
    WMMA_BF16(ah0, bh1, acc01); WMMA_BF16(ah0, bl1, acc01); WMMA_BF16(al0, bh1, acc01);
    WMMA_BF16(ah1, bh0, acc10); WMMA_BF16(ah1, bl0, acc10); WMMA_BF16(al1, bh0, acc10);
    WMMA_BF16(ah1, bh1, acc11); WMMA_BF16(ah1, bl1, acc11); WMMA_BF16(al1, bh1, acc11);
  }

  // epilogue: +bias, store. D layout: VGPR r -> M = r (lanes 0-15) / r+8 (lanes 16-31), N = lane&15
  const int   mrow = m0 + (lane >> 4) * 8;
  const int   ncol = n0 + (lane & 15);
  const float bn0  = bias[ncol];
  const float bn1  = bias[ncol + 16];
#pragma unroll
  for (int r = 0; r < 8; ++r) {
    P[(size_t)(mrow + r)      * D_ + ncol]      = acc00[r] + bn0;
    P[(size_t)(mrow + r)      * D_ + ncol + 16] = acc01[r] + bn1;
    P[(size_t)(mrow + 16 + r) * D_ + ncol]      = acc10[r] + bn0;
    P[(size_t)(mrow + 16 + r) * D_ + ncol + 16] = acc11[r] + bn1;
  }
}

// ---------------- 3) per-row inverse L2 norm (with 1e-12 clamp) ----------------
__global__ __launch_bounds__(256)
void rownorm_kernel(const float* __restrict__ P, float* __restrict__ inv) {
  __shared__ float red[256];
  const float* p = P + (size_t)blockIdx.x * D_;
  float s = 0.f;
  for (int d = threadIdx.x; d < D_; d += 256) { float v = p[d]; s += v * v; }
  red[threadIdx.x] = s;  __syncthreads();
  for (int off = 128; off > 0; off >>= 1) {
    if (threadIdx.x < off) red[threadIdx.x] += red[threadIdx.x + off];
    __syncthreads();
  }
  if (threadIdx.x == 0)
    inv[blockIdx.x] = 1.0f / fmaxf(sqrtf(red[0]), 1e-12f);
}

// ---------------- 4) attn = cos-sim, mask, top-5 indices ----------------
__global__ __launch_bounds__(256)
void attn_topk_kernel(const float* __restrict__ P, const float* __restrict__ inv,
                      const int* __restrict__ his_mask, int* __restrict__ idx_out) {
  __shared__ float crow[D_];
  __shared__ float red[256];
  __shared__ float attn_s[H_];
  const int bc = blockIdx.x;          // b*C + c
  const int b  = bc / C_;
  const float* cp   = P + (size_t)bc * D_;
  const float  invc = inv[bc];

  for (int d = threadIdx.x; d < D_; d += 256) crow[d] = cp[d];
  __syncthreads();

  for (int h = 0; h < H_; ++h) {
    const int hrow = B_ * C_ + b * H_ + h;
    const float* hp = P + (size_t)hrow * D_;
    float s = 0.f;
    for (int d = threadIdx.x; d < D_; d += 256) s += crow[d] * hp[d];
    red[threadIdx.x] = s;  __syncthreads();
    for (int off = 128; off > 0; off >>= 1) {
      if (threadIdx.x < off) red[threadIdx.x] += red[threadIdx.x + off];
      __syncthreads();
    }
    if (threadIdx.x == 0) {
      bool valid = (his_mask[b * H_ + h] != 0) || (h < KSEL);
      attn_s[h] = valid ? (red[0] * invc * inv[hrow]) : -__builtin_inff();
    }
    __syncthreads();
  }

  if (threadIdx.x == 0) {             // top-5, ties -> lowest index (matches lax.top_k)
    for (int kk = 0; kk < KSEL; ++kk) {
      float best = -__builtin_inff(); int bi = 0;
      for (int h = 0; h < H_; ++h)
        if (attn_s[h] > best) { best = attn_s[h]; bi = h; }
      idx_out[bc * KSEL + kk] = bi;
      attn_s[bi] = -__builtin_inff();
    }
  }
}

// ---------------- 5) bandwidth-bound gather (the roofline term) ----------------
__global__ __launch_bounds__(256)
void gather_kernel(const float* __restrict__ emb, const int* __restrict__ amask,
                   const int* __restrict__ idx, float* __restrict__ out_emb,
                   float* __restrict__ out_mask) {
  const int bck = blockIdx.x;                 // ((b*C + c)*K + k)
  const int bc  = bck / KSEL;
  const int b   = bc / C_;
  const int hidx = idx[bck];

  const f32x4* src = (const f32x4*)(emb + ((size_t)b * H_ + hidx) * (size_t)(S_ * E_));
  f32x4*       dst = (f32x4*)(out_emb + (size_t)bck * (size_t)(S_ * E_));
  const int nv = (S_ * E_) / 4;               // 6144 float4s (96 KB tile)
  for (int i = threadIdx.x; i < nv; i += 256) dst[i] = src[i];

  if (threadIdx.x < S_)
    out_mask[(size_t)bck * S_ + threadIdx.x] =
        (float)amask[((size_t)b * H_ + hidx) * S_ + threadIdx.x];
}

// ---------------- launcher ----------------
extern "C" void kernel_launch(void* const* d_in, const int* in_sizes, int n_in,
                              void* d_out, int out_size, void* d_ws, size_t ws_size,
                              hipStream_t stream) {
  const float* cdd   = (const float*)d_in[0];   // [B,C,D]
  const float* his   = (const float*)d_in[1];   // [B,H,D]
  const float* emb   = (const float*)d_in[2];   // [B,H,S,E]
  const int*   amask = (const int*)  d_in[3];   // [B,H,S]
  const int*   hmask = (const int*)  d_in[4];   // [B,H,1]
  const float* W     = (const float*)d_in[5];   // [D,D]
  const float* bias  = (const float*)d_in[6];   // [D]

  // workspace layout (256-B aligned offsets)
  char* ws = (char*)d_ws;
  uint16_t* Xh   = (uint16_t*)(ws + 0);          // 3520*768 bf16
  uint16_t* Xl   = (uint16_t*)(ws + 5406720);
  uint16_t* Wh   = (uint16_t*)(ws + 10813440);   // 768*768 bf16
  uint16_t* Wl   = (uint16_t*)(ws + 11993088);
  float*    P    = (float*)   (ws + 13172736);   // 3520*768 f32
  float*    invn = (float*)   (ws + 23986176);   // 3520 f32
  int*      idx  = (int*)     (ws + 24000512);   // 1600 int

  const int n_cdd = B_ * C_ * D_;   // 245760
  const int n_his = B_ * H_ * D_;   // 2457600
  const int n_w   = D_ * D_;        // 589824

  split_bf16_kernel<<<(n_cdd + 255) / 256, 256, 0, stream>>>(cdd, Xh, Xl, n_cdd);
  split_bf16_kernel<<<(n_his + 255) / 256, 256, 0, stream>>>(his, Xh + n_cdd, Xl + n_cdd, n_his);
  split_bf16_kernel<<<(n_w   + 255) / 256, 256, 0, stream>>>(W, Wh, Wl, n_w);

  proj_wmma_kernel<<<(MT_ * NT_) / 4, 128, 0, stream>>>(Xh, Xl, Wh, Wl, bias, P);
  rownorm_kernel<<<MROWS, 256, 0, stream>>>(P, invn);
  attn_topk_kernel<<<B_ * C_, 256, 0, stream>>>(P, invn, hmask, idx);

  float* out_emb  = (float*)d_out;
  float* out_mask = out_emb + (size_t)B_ * C_ * KSEL * S_ * E_;   // second tuple output
  gather_kernel<<<B_ * C_ * KSEL, 256, 0, stream>>>(emb, amask, idx, out_emb, out_mask);
}
